// SOPL_Layer_66511863546340
// MI455X (gfx1250) — matrix operand order repaired and checked by reading, not compiled, not run on hardware
//
#include <hip/hip_runtime.h>
#include <hip/hip_bf16.h>

// ---------------------------------------------------------------------------
// Soft k-means (SOPL), 3 iterations, WMMA pipeline for gfx1250 (MI455X).
//   N=200000, K=512, D=128, beta=1e-3.
// Softmax is shift-invariant per row => x_sq cancels; only q = 2*x.c - |c|^2
// matters.  Per launch: convert x once to bf16 (row-major xb + transposed
// xbT).  Per iteration:
//   prep   : cb = bf16(2*centers), c_sq
//   pass1  : q via WMMA (A from xb, B from cb, both global) -> per-row
//            (max, sumexp) stats
//   pass2  : per (16-center tile, N-chunk): recompute q, w = exp/s element-
//            wise, wT -> LDS in A-layout, num += wT*x via WMMA (B from xbT).
//            64-row rounds: waves 0-3 produce w, all 8 waves run 2 WMMA each.
//            Deterministic partials, no float atomics.
//   update : centers = num/(den+eps)
// ---------------------------------------------------------------------------

typedef __attribute__((ext_vector_type(16))) __bf16 v16bf;
typedef __attribute__((ext_vector_type(8)))  __bf16 v8bf;
typedef __attribute__((ext_vector_type(8)))  float  v8f;

#define N_PTS   200000
#define K_C     512
#define D_DIM   128
#define N_ITERS 3
#define BETA_INV 1000.0f
#define EPS_F    1e-10f
#define NCHUNK   25          // N-chunks for pass2 partials
#define GROUPS_PER_CHUNK 125 // 64-row rounds per chunk: 25*125*64 == 200000

// --- WMMA fragment loaders (layouts per CDNA5 ISA 7.12.2, bf16) -------------

// A-matrix 16x32 bf16: lane l<16 -> M=l, K={0..7,16..23}; lane>=16 -> M=l-16,
// K={8..15,24..31}.  `tile` points at (M=0,K=0), row-major, rowStride elems.
__device__ __forceinline__ v16bf load_A_16x32(const __bf16* tile, int rowStride, int lane) {
  const int m    = lane & 15;
  const int half = lane >> 4;
  const __bf16* p = tile + m * rowStride + half * 8;
  v8bf lo = *(const v8bf*)(p);                // K = half*8 ..
  v8bf hi = *(const v8bf*)(p + 16);           // K = 16+half*8 ..
  return __builtin_shufflevector(lo, hi, 0,1,2,3,4,5,6,7,8,9,10,11,12,13,14,15);
}

// B-matrix 32x16 bf16 with B[k][n] == srcT[n][k] (srcT row-major, rowStride
// elems): lane n<16 holds K={0..7,16..23} of column n; lane n+16 holds K={8..15,24..31}.
__device__ __forceinline__ v16bf load_B_32x16(const __bf16* srcT, int rowStride, int lane) {
  const int n    = lane & 15;
  const int koff = (lane >> 4) * 8;
  const __bf16* p = srcT + n * rowStride + koff;
  v8bf lo = *(const v8bf*)(p);
  v8bf hi = *(const v8bf*)(p + 16);
  return __builtin_shufflevector(lo, hi, 0,1,2,3,4,5,6,7,8,9,10,11,12,13,14,15);
}

// --- Kernel 0: one-time x -> bf16 (row-major xb, transposed xbT) ------------
// 64x64 tiles through LDS so both outputs store coalesced 32B runs.
__global__ void __launch_bounds__(256)
convert_x(const float* __restrict__ x, __bf16* __restrict__ xb,
          __bf16* __restrict__ xbT) {
  __shared__ __bf16 t[64][66];               // +2 pad vs bank conflicts
  const int n0 = blockIdx.x * 64;            // 3125 blocks
  const int d0 = blockIdx.y * 64;            // 2 blocks
  const int tid = threadIdx.x;
  { // load 64x64 f32: thread -> 16 contiguous floats of one row
    const int r = tid >> 2, dc = (tid & 3) * 16;
    const float4* src = (const float4*)(x + (size_t)(n0 + r) * D_DIM + d0 + dc);
    v8bf a, b;
    #pragma unroll
    for (int v = 0; v < 4; ++v) {
      float4 f = src[v];
      __bf16 c0 = (__bf16)f.x, c1 = (__bf16)f.y, c2 = (__bf16)f.z, c3 = (__bf16)f.w;
      t[r][dc + v*4 + 0] = c0; t[r][dc + v*4 + 1] = c1;
      t[r][dc + v*4 + 2] = c2; t[r][dc + v*4 + 3] = c3;
      v8bf& dst = (v < 2) ? a : b;
      dst[(v & 1)*4 + 0] = c0; dst[(v & 1)*4 + 1] = c1;
      dst[(v & 1)*4 + 2] = c2; dst[(v & 1)*4 + 3] = c3;
    }
    v8bf* out = (v8bf*)(xb + (size_t)(n0 + r) * D_DIM + d0 + dc);
    out[0] = a; out[1] = b;
  }
  __syncthreads();
  { // write xbT: thread -> 16 contiguous n of one d-row
    const int d = tid >> 2, nc = (tid & 3) * 16;
    v8bf a, b;
    #pragma unroll
    for (int i = 0; i < 16; ++i) {
      __bf16 v = t[nc + i][d];
      if (i < 8) a[i] = v; else b[i - 8] = v;
    }
    v8bf* out = (v8bf*)(xbT + (size_t)(d0 + d) * N_PTS + n0 + nc);
    out[0] = a; out[1] = b;
  }
}

// --- Kernel 1: prep  cb = bf16(2*centers), c_sq[k] = |c_k|^2 ----------------
__global__ void __launch_bounds__(128)
prep_kernel(const float* __restrict__ centers, __bf16* __restrict__ cb,
            float* __restrict__ c_sq) {
  __shared__ float red[128];
  const int k = blockIdx.x, d = threadIdx.x;
  const float c = centers[k * D_DIM + d];
  cb[k * D_DIM + d] = (__bf16)(2.0f * c);
  red[d] = c * c;
  __syncthreads();
  for (int s = 64; s > 0; s >>= 1) {
    if (d < s) red[d] += red[d + s];
    __syncthreads();
  }
  if (d == 0) c_sq[k] = red[0];
}

// --- Kernel 2: pass1  per-row softmax stats (qmax, sumexp) ------------------
// Block = 256 threads (8 waves), 16 rows.  A fragments held in registers and
// reused across the wave's 4 center tiles; no x staging.
__global__ void __launch_bounds__(256)
pass1_stats(const __bf16* __restrict__ xb, const __bf16* __restrict__ cb,
            const float* __restrict__ c_sq, float* __restrict__ stats) {
  __shared__ float q[16][K_C];        // 32 KB
  const int tid  = threadIdx.x;
  const int lane = tid & 31;
  const int wave = tid >> 5;
  const int r0   = blockIdx.x * 16;   // 12500 * 16 == 200000 exactly

  v16bf a0 = load_A_16x32(xb + (size_t)r0 * D_DIM +  0, D_DIM, lane);
  v16bf a1 = load_A_16x32(xb + (size_t)r0 * D_DIM + 32, D_DIM, lane);
  v16bf a2 = load_A_16x32(xb + (size_t)r0 * D_DIM + 64, D_DIM, lane);
  v16bf a3 = load_A_16x32(xb + (size_t)r0 * D_DIM + 96, D_DIM, lane);

  #pragma unroll
  for (int t = 0; t < 4; ++t) {
    const int c0 = (wave * 4 + t) * 16;
    const float negc = -c_sq[c0 + (lane & 15)];
    v8f acc = { negc, negc, negc, negc, negc, negc, negc, negc };
    acc = __builtin_amdgcn_wmma_f32_16x16x32_bf16(false, a0, false,
            load_B_32x16(cb + (size_t)c0 * D_DIM +  0, D_DIM, lane), (short)0, acc, false, false);
    acc = __builtin_amdgcn_wmma_f32_16x16x32_bf16(false, a1, false,
            load_B_32x16(cb + (size_t)c0 * D_DIM + 32, D_DIM, lane), (short)0, acc, false, false);
    acc = __builtin_amdgcn_wmma_f32_16x16x32_bf16(false, a2, false,
            load_B_32x16(cb + (size_t)c0 * D_DIM + 64, D_DIM, lane), (short)0, acc, false, false);
    acc = __builtin_amdgcn_wmma_f32_16x16x32_bf16(false, a3, false,
            load_B_32x16(cb + (size_t)c0 * D_DIM + 96, D_DIM, lane), (short)0, acc, false, false);
    const int col  = c0 + (lane & 15);
    const int rAdd = (lane >> 4) * 8;
    #pragma unroll
    for (int j = 0; j < 8; ++j) q[rAdd + j][col] = acc[j];
  }
  __syncthreads();

  // per-row reduction: 16 lanes per row (contiguous within a wave)
  const int r = tid >> 4, sub = tid & 15;
  float m = -3.4e38f;
  #pragma unroll 4
  for (int i = 0; i < 32; ++i) m = fmaxf(m, q[r][sub + 16 * i]);
  #pragma unroll
  for (int w = 8; w >= 1; w >>= 1) m = fmaxf(m, __shfl_xor(m, w, 16));
  float s = 0.f;
  #pragma unroll 4
  for (int i = 0; i < 32; ++i) s += __expf((q[r][sub + 16 * i] - m) * BETA_INV);
  #pragma unroll
  for (int w = 8; w >= 1; w >>= 1) s += __shfl_xor(s, w, 16);
  if (sub == 0) {
    stats[(size_t)(r0 + r) * 2 + 0] = m;
    stats[(size_t)(r0 + r) * 2 + 1] = s;
  }
}

// --- Kernel 3: pass2  num/den partials per (k-tile, chunk) ------------------
// Block = 256 threads (8 waves).  64-row rounds: waves 0-3 recompute q for 16
// rows each and emit w^T (bf16, A-layout) to LDS; all 8 waves then run two
// WMMA accumulating their num[16c x 16d] tile over the 64 rows.
__global__ void __launch_bounds__(256)
pass2_accum(const __bf16* __restrict__ xb, const __bf16* __restrict__ xbT,
            const __bf16* __restrict__ cb, const float* __restrict__ c_sq,
            const float* __restrict__ stats,
            float* __restrict__ num_part, float* __restrict__ den_part) {
  __shared__ __bf16 wT[16][64];      // 2 KB: [center][local row]
  __shared__ float  den_lds[4][16];
  const int tid  = threadIdx.x;
  const int lane = tid & 31;
  const int wave = tid >> 5;
  const int chunk = blockIdx.x;      // [0, NCHUNK)
  const int kt    = blockIdx.y;      // [0, 32)
  const int c0    = kt * 16;
  const float negc = -c_sq[c0 + (lane & 15)];
  const float2* st2 = (const float2*)stats;

  v8f   numacc = { 0.f, 0.f, 0.f, 0.f, 0.f, 0.f, 0.f, 0.f };
  float denacc = 0.f;

  for (int g = 0; g < GROUPS_PER_CHUNK; ++g) {
    const int rowbase = (chunk * GROUPS_PER_CHUNK + g) * 64;
    __syncthreads();   // protect wT from previous round's GEMM2 readers

    if (wave < 4) { // q for rows [wave*16, +16) x 16 centers -> w^T
      const int rbase = rowbase + wave * 16;
      v8f acc = { negc, negc, negc, negc, negc, negc, negc, negc };
      #pragma unroll
      for (int kk = 0; kk < 4; ++kk) {
        v16bf a = load_A_16x32(xb + (size_t)rbase * D_DIM + kk * 32, D_DIM, lane);
        v16bf b = load_B_32x16(cb + (size_t)c0 * D_DIM + kk * 32, D_DIM, lane);
        acc = __builtin_amdgcn_wmma_f32_16x16x32_bf16(false, a, false, b,
                                                      (short)0, acc, false, false);
      }
      const int c    = lane & 15;
      const int rAdd = wave * 16 + ((lane >> 4) * 8);
      #pragma unroll
      for (int j = 0; j < 8; ++j) {
        const int rl = rAdd + j;                // local row in [0,64)
        const float2 ms = st2[rowbase + rl];    // (qmax, sumexp)
        const float wv = __expf((acc[j] - ms.x) * BETA_INV) / ms.y;
        denacc += wv;
        wT[c][rl] = (__bf16)wv;
      }
    }
    __syncthreads();

    { // num[16c x 16d] += w^T(16x64) * x(64x16d); wave owns d-tile wave*16
      const __bf16* xTrow = xbT + (size_t)(wave * 16) * N_PTS + rowbase;
      v16bf a0 = load_A_16x32(&wT[0][0],  64, lane);
      v16bf b0 = load_B_32x16(xTrow,      N_PTS, lane);
      numacc = __builtin_amdgcn_wmma_f32_16x16x32_bf16(false, a0, false, b0,
                                                       (short)0, numacc, false, false);
      v16bf a1 = load_A_16x32(&wT[0][32], 64, lane);
      v16bf b1 = load_B_32x16(xTrow + 32, N_PTS, lane);
      numacc = __builtin_amdgcn_wmma_f32_16x16x32_bf16(false, a1, false, b1,
                                                       (short)0, numacc, false, false);
    }
  }

  // write num partial tile [16 centers][128 d] for this (kt, chunk)
  const int pb = kt * NCHUNK + chunk;
  {
    float* np = num_part + (size_t)pb * 16 * D_DIM;
    const int col  = wave * 16 + (lane & 15);
    const int rAdd = (lane >> 4) * 8;
    #pragma unroll
    for (int j = 0; j < 8; ++j) np[(size_t)(rAdd + j) * D_DIM + col] = numacc[j];
  }
  // den partial: lane c holds rows j, lane c+16 rows j+8 of its M-tile
  denacc += __shfl_xor(denacc, 16, 32);
  if (wave < 4 && lane < 16) den_lds[wave][lane] = denacc;
  __syncthreads();
  if (tid < 16)
    den_part[(size_t)pb * 16 + tid] =
        (den_lds[0][tid] + den_lds[1][tid]) + (den_lds[2][tid] + den_lds[3][tid]);
}

// --- Kernel 4: reduce partials, centers = num/(den+eps) ---------------------
__global__ void __launch_bounds__(128)
update_kernel(const float* __restrict__ num_part, const float* __restrict__ den_part,
              float* __restrict__ centers_out) {
  const int k = blockIdx.x, d = threadIdx.x;
  const int kt = k >> 4, r = k & 15;
  float nsum = 0.f, dsum = 0.f;
  for (int c = 0; c < NCHUNK; ++c) {
    nsum += num_part[((size_t)(kt * NCHUNK + c) * 16 + r) * D_DIM + d];
    dsum += den_part[(size_t)(kt * NCHUNK + c) * 16 + r];
  }
  centers_out[k * D_DIM + d] = nsum / (dsum + EPS_F);
}

// ---------------------------------------------------------------------------
extern "C" void kernel_launch(void* const* d_in, const int* in_sizes, int n_in,
                              void* d_out, int out_size, void* d_ws, size_t ws_size,
                              hipStream_t stream) {
  (void)in_sizes; (void)n_in; (void)out_size; (void)ws_size;
  const float* x        = (const float*)d_in[0];
  const float* centers0 = (const float*)d_in[1];
  float*       out      = (float*)d_out;

  // workspace carve-out (~111 MB total)
  char*  ws  = (char*)d_ws;
  size_t off = 0;
  auto carve = [&](size_t bytes) -> void* {
    void* p = ws + off;
    off += (bytes + 255) & ~(size_t)255;
    return p;
  };
  __bf16* cb       = (__bf16*)carve((size_t)K_C * D_DIM * sizeof(__bf16));
  float*  c_sq     = (float*) carve((size_t)K_C * sizeof(float));
  float*  stats    = (float*) carve((size_t)N_PTS * 2 * sizeof(float));
  float*  num_part = (float*) carve((size_t)32 * NCHUNK * 16 * D_DIM * sizeof(float));
  float*  den_part = (float*) carve((size_t)32 * NCHUNK * 16 * sizeof(float));
  __bf16* xb       = (__bf16*)carve((size_t)N_PTS * D_DIM * sizeof(__bf16));
  __bf16* xbT      = (__bf16*)carve((size_t)D_DIM * N_PTS * sizeof(__bf16));

  // one-time bf16 conversion of x (iteration-invariant)
  convert_x<<<dim3(N_PTS / 64, D_DIM / 64), 256, 0, stream>>>(x, xb, xbT);

  for (int it = 0; it < N_ITERS; ++it) {
    const float* csrc = (it == 0) ? centers0 : out;
    prep_kernel <<<K_C, D_DIM, 0, stream>>>(csrc, cb, c_sq);
    pass1_stats <<<N_PTS / 16, 256, 0, stream>>>(xb, cb, c_sq, stats);
    pass2_accum <<<dim3(NCHUNK, K_C / 16), 256, 0, stream>>>(xb, xbT, cb, c_sq,
                                                             stats, num_part, den_part);
    update_kernel<<<K_C, D_DIM, 0, stream>>>(num_part, den_part, out);
  }
}
